// SingleMambaBlock_78683800862958
// MI455X (gfx1250) — matrix operand order repaired and checked by reading, not compiled, not run on hardware
//
#include <hip/hip_runtime.h>

// ---- problem dims ----
#define B_DIM      2
#define L_DIM      4096
#define D_MODEL    256
#define D_INNER    512
#define D_STATE    8
#define DT_RANK    16
#define NROW       (B_DIM * L_DIM)      // 8192
#define NCH        16                    // scan chunks per sequence
#define LC         (L_DIM / NCH)         // 256

typedef __attribute__((ext_vector_type(16))) _Float16 v16h;
typedef __attribute__((ext_vector_type(8)))  _Float16 v8h;
typedef __attribute__((ext_vector_type(8)))  float    v8f;

__device__ __forceinline__ float silu_f(float x) {
    return x * (1.0f / (1.0f + __expf(-x)));
}

// ---------------- weight cast + transpose: W[K][N] fp32 -> Wt[N][K] f16 ----------------
__global__ void castT_kernel(const float* __restrict__ W, _Float16* __restrict__ Wt,
                             int K, int N) {
    int idx = blockIdx.x * blockDim.x + threadIdx.x;
    if (idx >= K * N) return;
    int n = idx % N, k = idx / N;
    Wt[(size_t)n * K + k] = (_Float16)W[(size_t)k * N + n];
}

// ---------------- layernorm (one 256-wide row per block) -> f16 ----------------
__global__ void ln_cast_kernel(const float* __restrict__ x, const float* __restrict__ g,
                               const float* __restrict__ bta, _Float16* __restrict__ h16) {
    int row = blockIdx.x;
    int tid = threadIdx.x;                       // 256 threads == D_MODEL
    __shared__ float red[D_MODEL];
    float v = x[(size_t)row * D_MODEL + tid];
    red[tid] = v; __syncthreads();
    for (int s = 128; s > 0; s >>= 1) { if (tid < s) red[tid] += red[tid + s]; __syncthreads(); }
    float mu = red[0] * (1.0f / D_MODEL);
    __syncthreads();
    float dv = v - mu;
    red[tid] = dv * dv; __syncthreads();
    for (int s = 128; s > 0; s >>= 1) { if (tid < s) red[tid] += red[tid + s]; __syncthreads(); }
    float var = red[0] * (1.0f / D_MODEL);
    float o = dv * rsqrtf(var + 1e-5f) * g[tid] + bta[tid];
    h16[(size_t)row * D_MODEL + tid] = (_Float16)o;
}

// ---------------- WMMA GEMM: C[M,N] = A[M,K] * Bt[N,K]^T (+ optional skip) ----------------
// One wave per 16x16 output tile. A, Bt f16 row-major; K multiple of 32.
__global__ void wmma_gemm_f16(const _Float16* __restrict__ A,
                              const _Float16* __restrict__ Bt,
                              float* __restrict__ C,
                              const float* __restrict__ skip,
                              int M, int N, int K) {
    int wave = (blockIdx.x * blockDim.x + threadIdx.x) >> 5;   // wave32
    int lane = threadIdx.x & 31;
    int tilesN = N >> 4;
    int tilesM = M >> 4;
    if (wave >= tilesM * tilesN) return;                       // wave-uniform exit
    int tm = wave / tilesN, tn = wave % tilesN;
    int half = lane >> 4, r = lane & 15;

    const _Float16* Arow = A  + (size_t)(tm * 16 + r) * K;     // A: row M = tm*16+r
    const _Float16* Brow = Bt + (size_t)(tn * 16 + r) * K;     // B col N = tn*16+r

    v8f acc = {};
    for (int k0 = 0; k0 < K; k0 += 32) {
        // A 16x32 layout: lane-half 0 -> K {0..7, 16..23}; half 1 -> K {8..15, 24..31}
        v8h alo = *(const v8h*)(Arow + k0 + half * 8);
        v8h ahi = *(const v8h*)(Arow + k0 + 16 + half * 8);
        // B 32x16 layout: lane-half 0 -> K {0..7, 16..23}; half 1 -> K {8..15, 24..31}
        v8h blo = *(const v8h*)(Brow + k0 + half * 8);
        v8h bhi = *(const v8h*)(Brow + k0 + 16 + half * 8);
        v16h a = __builtin_shufflevector(alo, ahi, 0,1,2,3,4,5,6,7,8,9,10,11,12,13,14,15);
        v16h b = __builtin_shufflevector(blo, bhi, 0,1,2,3,4,5,6,7,8,9,10,11,12,13,14,15);
        acc = __builtin_amdgcn_wmma_f32_16x16x32_f16(false, a, false, b, (short)0, acc,
                                                     false, false);
    }
    // C/D layout: VGPR v -> M = v + 8*half, N = r
    #pragma unroll
    for (int v = 0; v < 8; ++v) {
        size_t row = (size_t)(tm * 16 + v + 8 * half);
        size_t col = (size_t)(tn * 16 + r);
        float o = acc[v];
        if (skip) o += skip[row * N + col];
        C[row * N + col] = o;
    }
}

// ---------------- causal depthwise conv (k=4) + SiLU; emits fp32 and f16 ----------------
__global__ void conv_silu_kernel(const float* __restrict__ xz, const float* __restrict__ cw,
                                 const float* __restrict__ cb,
                                 float* __restrict__ xc, _Float16* __restrict__ xc16) {
    int idx = blockIdx.x * blockDim.x + threadIdx.x;           // over NROW*D_INNER
    if (idx >= NROW * D_INNER) return;
    int d  = idx & (D_INNER - 1);
    int bl = idx >> 9;
    int l  = bl & (L_DIM - 1);
    int b  = bl >> 12;
    float acc = cb[d];
    #pragma unroll
    for (int k = 0; k < 4; ++k) {
        int ls = l - 3 + k;
        if (ls >= 0)
            acc += cw[d * 4 + k] * xz[((size_t)(b * L_DIM + ls)) * (2 * D_INNER) + d];
    }
    float s = silu_f(acc);
    xc[idx]   = s;
    xc16[idx] = (_Float16)s;
}

// ---------------- dt = softplus(x_dbl[:, :16] @ W_dt + dt_bias) ----------------
__global__ void dt_kernel(const float* __restrict__ xdbl, const float* __restrict__ Wdt,
                          const float* __restrict__ dtb, float* __restrict__ dt) {
    int idx = blockIdx.x * blockDim.x + threadIdx.x;           // over NROW*D_INNER
    if (idx >= NROW * D_INNER) return;
    int d = idx & (D_INNER - 1);
    int row = idx >> 9;
    float acc = dtb[d];
    #pragma unroll
    for (int j = 0; j < DT_RANK; ++j)
        acc += xdbl[(size_t)row * 32 + j] * Wdt[(size_t)j * D_INNER + d];
    dt[idx] = (acc > 20.0f) ? acc : log1pf(__expf(acc));
}

// ---------------- chunked selective scan: pass 1 (local scan, save chunk summaries) ----------
__global__ void scan_pass1(const float* __restrict__ dt, const float* __restrict__ xc,
                           const float* __restrict__ xdbl, const float* __restrict__ A_log,
                           float* __restrict__ prodA, float* __restrict__ hend) {
    int t = blockIdx.x * blockDim.x + threadIdx.x;             // B*NCH*D_INNER
    if (t >= B_DIM * NCH * D_INNER) return;
    int d  = t & (D_INNER - 1);
    int bc = t >> 9;
    int c  = bc & (NCH - 1);
    int b  = bc >> 4;
    float nA[D_STATE];
    #pragma unroll
    for (int n = 0; n < D_STATE; ++n) nA[n] = -__expf(A_log[d * D_STATE + n]);
    float h[D_STATE], p[D_STATE];
    #pragma unroll
    for (int n = 0; n < D_STATE; ++n) { h[n] = 0.0f; p[n] = 1.0f; }
    int l0 = c * LC;
    for (int l = l0; l < l0 + LC; ++l) {
        size_t row = (size_t)b * L_DIM + l;
        float dtv = dt[row * D_INNER + d];
        float xv  = xc[row * D_INNER + d];
        const float* Bm = xdbl + row * 32 + DT_RANK;
        #pragma unroll
        for (int n = 0; n < D_STATE; ++n) {
            float dA = __expf(dtv * nA[n]);
            h[n] = dA * h[n] + dtv * Bm[n] * xv;
            p[n] *= dA;
        }
    }
    size_t base = ((size_t)(b * NCH + c) * D_STATE) * D_INNER + d;
    #pragma unroll
    for (int n = 0; n < D_STATE; ++n) {
        prodA[base + (size_t)n * D_INNER] = p[n];
        hend [base + (size_t)n * D_INNER] = h[n];
    }
}

// ---------------- chunked scan: pass 2 (serial carry across chunks, per channel) ----------
__global__ void scan_carry(const float* __restrict__ prodA, const float* __restrict__ hend,
                           float* __restrict__ carry) {
    int t = blockIdx.x * blockDim.x + threadIdx.x;             // B*D_INNER
    if (t >= B_DIM * D_INNER) return;
    int d = t & (D_INNER - 1);
    int b = t >> 9;
    float h[D_STATE];
    #pragma unroll
    for (int n = 0; n < D_STATE; ++n) h[n] = 0.0f;
    for (int c = 0; c < NCH; ++c) {
        size_t base = ((size_t)(b * NCH + c) * D_STATE) * D_INNER + d;
        #pragma unroll
        for (int n = 0; n < D_STATE; ++n) {
            carry[base + (size_t)n * D_INNER] = h[n];
            h[n] = prodA[base + (size_t)n * D_INNER] * h[n]
                 + hend [base + (size_t)n * D_INNER];
        }
    }
}

// ---------------- chunked scan: pass 3 (replay with carry, fuse y = h.C + xc*D, gate) ------
__global__ void scan_pass3(const float* __restrict__ dt, const float* __restrict__ xc,
                           const float* __restrict__ xdbl, const float* __restrict__ A_log,
                           const float* __restrict__ Dp, const float* __restrict__ xz,
                           const float* __restrict__ carry, _Float16* __restrict__ y16) {
    int t = blockIdx.x * blockDim.x + threadIdx.x;             // B*NCH*D_INNER
    if (t >= B_DIM * NCH * D_INNER) return;
    int d  = t & (D_INNER - 1);
    int bc = t >> 9;
    int c  = bc & (NCH - 1);
    int b  = bc >> 4;
    float nA[D_STATE];
    #pragma unroll
    for (int n = 0; n < D_STATE; ++n) nA[n] = -__expf(A_log[d * D_STATE + n]);
    float h[D_STATE];
    size_t cb = ((size_t)(b * NCH + c) * D_STATE) * D_INNER + d;
    #pragma unroll
    for (int n = 0; n < D_STATE; ++n) h[n] = carry[cb + (size_t)n * D_INNER];
    float Dd = Dp[d];
    int l0 = c * LC;
    for (int l = l0; l < l0 + LC; ++l) {
        size_t row = (size_t)b * L_DIM + l;
        float dtv = dt[row * D_INNER + d];
        float xv  = xc[row * D_INNER + d];
        const float* Bm = xdbl + row * 32 + DT_RANK;
        const float* Cm = xdbl + row * 32 + DT_RANK + D_STATE;
        float y = 0.0f;
        #pragma unroll
        for (int n = 0; n < D_STATE; ++n) {
            float dA = __expf(dtv * nA[n]);
            h[n] = dA * h[n] + dtv * Bm[n] * xv;
            y += h[n] * Cm[n];
        }
        y += xv * Dd;
        float z = xz[row * (2 * D_INNER) + D_INNER + d];
        y16[row * D_INNER + d] = (_Float16)(y * silu_f(z));
    }
}

// -------------------------------------------------------------------------------------------
extern "C" void kernel_launch(void* const* d_in, const int* in_sizes, int n_in,
                              void* d_out, int out_size, void* d_ws, size_t ws_size,
                              hipStream_t stream) {
    const float* x       = (const float*)d_in[0];
    const float* ln_g    = (const float*)d_in[1];
    const float* ln_b    = (const float*)d_in[2];
    const float* W_in    = (const float*)d_in[3];
    const float* conv_w  = (const float*)d_in[4];
    const float* conv_b  = (const float*)d_in[5];
    const float* W_x     = (const float*)d_in[6];
    const float* W_dt    = (const float*)d_in[7];
    const float* dt_bias = (const float*)d_in[8];
    const float* A_log   = (const float*)d_in[9];
    const float* Dp      = (const float*)d_in[10];
    const float* W_out   = (const float*)d_in[11];
    float* out = (float*)d_out;

    char* w = (char*)d_ws;
    auto alloc = [&](size_t bytes) -> void* {
        void* p = (void*)w;
        w += (bytes + 255) & ~(size_t)255;
        return p;
    };
    _Float16* h16   = (_Float16*)alloc((size_t)NROW * D_MODEL * 2);
    float*    xz    = (float*)   alloc((size_t)NROW * 2 * D_INNER * 4);
    float*    xc    = (float*)   alloc((size_t)NROW * D_INNER * 4);
    _Float16* xc16  = (_Float16*)alloc((size_t)NROW * D_INNER * 2);
    float*    xdbl  = (float*)   alloc((size_t)NROW * 32 * 4);
    float*    dt    = (float*)   alloc((size_t)NROW * D_INNER * 4);
    _Float16* y16   = (_Float16*)alloc((size_t)NROW * D_INNER * 2);
    size_t summ = (size_t)B_DIM * NCH * D_STATE * D_INNER * 4;
    float*    prodA = (float*)   alloc(summ);
    float*    hend  = (float*)   alloc(summ);
    float*    carry = (float*)   alloc(summ);
    _Float16* WinT  = (_Float16*)alloc((size_t)2 * D_INNER * D_MODEL * 2);
    _Float16* WxT   = (_Float16*)alloc((size_t)32 * D_INNER * 2);
    _Float16* WoutT = (_Float16*)alloc((size_t)D_MODEL * D_INNER * 2);

    // 1) cast + transpose weights to f16 [N][K]
    castT_kernel<<<(D_MODEL * 2 * D_INNER + 255) / 256, 256, 0, stream>>>(W_in, WinT, D_MODEL, 2 * D_INNER);
    castT_kernel<<<(D_INNER * 32 + 255) / 256, 256, 0, stream>>>(W_x, WxT, D_INNER, DT_RANK + 2 * D_STATE);
    castT_kernel<<<(D_INNER * D_MODEL + 255) / 256, 256, 0, stream>>>(W_out, WoutT, D_INNER, D_MODEL);

    // 2) layernorm -> f16
    ln_cast_kernel<<<NROW, D_MODEL, 0, stream>>>(x, ln_g, ln_b, h16);

    // 3) xz = h @ W_in   (M=8192, N=1024, K=256)
    {
        int tiles = (NROW / 16) * ((2 * D_INNER) / 16);
        wmma_gemm_f16<<<(tiles * 32 + 255) / 256, 256, 0, stream>>>(
            h16, WinT, xz, nullptr, NROW, 2 * D_INNER, D_MODEL);
    }

    // 4) causal depthwise conv + SiLU
    conv_silu_kernel<<<(NROW * D_INNER + 255) / 256, 256, 0, stream>>>(xz, conv_w, conv_b, xc, xc16);

    // 5) x_dbl = xc @ W_x   (M=8192, N=32, K=512)
    {
        int tiles = (NROW / 16) * (32 / 16);
        wmma_gemm_f16<<<(tiles * 32 + 255) / 256, 256, 0, stream>>>(
            xc16, WxT, xdbl, nullptr, NROW, 32, D_INNER);
    }

    // 6) dt = softplus(x_dbl[:, :16] @ W_dt + dt_bias)
    dt_kernel<<<(NROW * D_INNER + 255) / 256, 256, 0, stream>>>(xdbl, W_dt, dt_bias, dt);

    // 7-9) chunked selective scan (fused discretization, no deltaA/deltaBx materialization)
    scan_pass1<<<(B_DIM * NCH * D_INNER + 255) / 256, 256, 0, stream>>>(dt, xc, xdbl, A_log, prodA, hend);
    scan_carry<<<(B_DIM * D_INNER + 255) / 256, 256, 0, stream>>>(prodA, hend, carry);
    scan_pass3<<<(B_DIM * NCH * D_INNER + 255) / 256, 256, 0, stream>>>(dt, xc, xdbl, A_log, Dp, xz, carry, y16);

    // 10) out = y @ W_out + skip   (M=8192, N=256, K=512)
    {
        int tiles = (NROW / 16) * (D_MODEL / 16);
        wmma_gemm_f16<<<(tiles * 32 + 255) / 256, 256, 0, stream>>>(
            y16, WoutT, out, x, NROW, D_MODEL, D_INNER);
    }
    (void)in_sizes; (void)n_in; (void)out_size; (void)ws_size;
}
